// GraphAttentionLayer_489626272420
// MI455X (gfx1250) — matrix operand order repaired and checked by reading, not compiled, not run on hardware
//
#include <hip/hip_runtime.h>
#include <hip/hip_bf16.h>

typedef __attribute__((ext_vector_type(16))) _Float16 v16h;
typedef __attribute__((ext_vector_type(8)))  float    v8f;

union AF { v16h v; _Float16 h[16]; };
union CF { v8f  v; float    f[8];  };

#define Bq 8
#define Sq 512
#define Dq 1024
#define Hq 8
#define DA 128

// ---------------------------------------------------------------------------
// Kernel 1: xp[4096,1024] (f16) = X[4096,1024] (f32) @ W[1024,1024] (f32)
// Tile 128x64, BK=32, 8 waves; each wave -> 16(M) x 64(N) via 4 accumulators.
// ---------------------------------------------------------------------------
__global__ __launch_bounds__(256) void proj_gemm(const float* __restrict__ X,
                                                 const float* __restrict__ W,
                                                 _Float16* __restrict__ XP) {
  __shared__ _Float16 As[128 * 32];   // [m][k]
  __shared__ _Float16 Bs[64 * 32];    // [n][k]  (transposed so K-pairs are contiguous)
  const int t    = threadIdx.x;
  const int wave = t >> 5;
  const int lane = t & 31;
  const int m0 = blockIdx.y * 128;
  const int n0 = blockIdx.x * 64;

  CF acc[4];
  acc[0].v = (v8f){}; acc[1].v = (v8f){}; acc[2].v = (v8f){}; acc[3].v = (v8f){};

  for (int k0 = 0; k0 < Dq; k0 += 32) {
    __syncthreads();
    // stage A: 128x32 f32 -> f16 ; 4 float4 per thread
    {
      const int kq   = (t & 7) * 4;
      const int mrow = t >> 3;           // 0..31
      #pragma unroll
      for (int i = 0; i < 4; ++i) {
        const int m = mrow + i * 32;
        const float4 f = *(const float4*)(X + (size_t)(m0 + m) * Dq + k0 + kq);
        _Float16* dst = As + m * 32 + kq;
        dst[0] = (_Float16)f.x; dst[1] = (_Float16)f.y;
        dst[2] = (_Float16)f.z; dst[3] = (_Float16)f.w;
      }
    }
    // stage B transposed: W rows k -> Bs[n][k]
    {
      const int nq = (t & 15) * 4;
      const int kr = t >> 4;             // 0..15
      #pragma unroll
      for (int i = 0; i < 2; ++i) {
        const int k = kr + i * 16;
        const float4 f = *(const float4*)(W + (size_t)(k0 + k) * Dq + n0 + nq);
        Bs[(nq + 0) * 32 + k] = (_Float16)f.x;
        Bs[(nq + 1) * 32 + k] = (_Float16)f.y;
        Bs[(nq + 2) * 32 + k] = (_Float16)f.z;
        Bs[(nq + 3) * 32 + k] = (_Float16)f.w;
      }
    }
    __syncthreads();

    const int Ml = lane & 15, half = lane >> 4;
    AF a;
    const _Float16* arow = As + (wave * 16 + Ml) * 32;
    #pragma unroll
    for (int e = 0; e < 16; ++e) {
      const int v = e >> 1;
      const int K = (v >> 2) * 16 + half * 8 + (v & 3) * 2 + (e & 1);
      a.h[e] = arow[K];
    }
    #pragma unroll
    for (int j = 0; j < 4; ++j) {
      AF bf;
      const _Float16* bcol = Bs + (j * 16 + Ml) * 32;
      #pragma unroll
      for (int e = 0; e < 16; ++e) {
        const int v = e >> 1;
        const int K = (v >> 2) * 16 + half * 8 + (v & 3) * 2 + (e & 1);
        bf.h[e] = bcol[K];
      }
      acc[j].v = __builtin_amdgcn_wmma_f32_16x16x32_f16(
          false, a.v, false, bf.v, (short)0, acc[j].v, false, false);
    }
  }

  // epilogue: C layout -> VGPR r: lanes0-15 M=r, lanes16-31 M=8+r; N = lane%16
  const int nl = lane & 15, half = lane >> 4;
  #pragma unroll
  for (int j = 0; j < 4; ++j)
    #pragma unroll
    for (int r = 0; r < 8; ++r) {
      const int m = wave * 16 + half * 8 + r;
      XP[(size_t)(m0 + m) * Dq + n0 + j * 16 + nl] = (_Float16)acc[j].f[r];
    }
}

// ---------------------------------------------------------------------------
// Kernel 2: s2[b,h,j] = dot(xp[b,j,h*128 : h*128+128], att_w[128:256])
// (s1 and att_b cancel inside the row softmax, so only s2 is needed)
// ---------------------------------------------------------------------------
__global__ __launch_bounds__(256) void s2_kernel(const _Float16* __restrict__ XP,
                                                 const float* __restrict__ attw,
                                                 float* __restrict__ S2) {
  const int tid = blockIdx.x * 256 + threadIdx.x;   // b*4096 + h*512 + j
  if (tid >= Bq * Hq * Sq) return;
  const int j = tid & 511;
  const int h = (tid >> 9) & 7;
  const int b = tid >> 12;
  const _Float16* row = XP + ((size_t)(b * Sq + j)) * Dq + h * DA;
  float s = 0.f;
  #pragma unroll 8
  for (int d = 0; d < DA; ++d) s += (float)row[d] * attw[DA + d];
  S2[tid] = s;
}

// ---------------------------------------------------------------------------
// Kernel 3: fused masked softmax + (att @ V) per (b,h, 32-row block).
// 8 waves laid out 2(M) x 4(N); each wave: 16x32 output = 2 accumulators.
// V tile staged with GLOBAL_LOAD_ASYNC_TO_LDS_B128 (ASYNCcnt path).
// ---------------------------------------------------------------------------
__global__ __launch_bounds__(256) void attn_kernel(const int* __restrict__ adj,
                                                   const float* __restrict__ mask,
                                                   const float* __restrict__ S2,
                                                   const _Float16* __restrict__ XP,
                                                   float* __restrict__ OUT) {
  __shared__ float    s2m[Sq];
  __shared__ float    rowm[32];
  __shared__ float    rowdinv[32];
  __shared__ int      rowuni[32];
  __shared__ _Float16 ptile[32 * 32];     // [i][j]
  __shared__ _Float16 vtile[32 * DA];     // [k][n]

  const int t    = threadIdx.x;
  const int iblk = blockIdx.x;
  const int h    = blockIdx.y;
  const int b    = blockIdx.z;
  const int i0   = iblk * 32;

  // column-masked s2 into LDS
  for (int j = t; j < Sq; j += 256)
    s2m[j] = (mask[b * Sq + j] > 0.f) ? S2[(b * Hq + h) * Sq + j] : -1e30f;
  __syncthreads();

  // per-row max & sum, 8 threads per row
  {
    const int r = t >> 3, sub = t & 7;
    const int i = i0 + r;
    const int* arow = adj + ((size_t)(b * Sq + i)) * Sq;
    const float rmask = mask[b * Sq + i];
    float m = -1e30f;
    for (int j = sub; j < Sq; j += 8)
      if (arow[j] > 0) m = fmaxf(m, s2m[j]);
    #pragma unroll
    for (int off = 4; off >= 1; off >>= 1)
      m = fmaxf(m, __shfl_xor(m, off, 8));
    float s = 0.f;
    if (m > -1e29f) {
      for (int j = sub; j < Sq; j += 8)
        if (arow[j] > 0) s += __expf(s2m[j] - m);
    }
    #pragma unroll
    for (int off = 4; off >= 1; off >>= 1)
      s += __shfl_xor(s, off, 8);
    if (sub == 0) {
      const int uni = (rmask <= 0.f) || (s <= 0.f) || (m <= -1e29f);
      rowuni[r]   = uni;
      rowm[r]     = m;
      rowdinv[r]  = uni ? (1.f / (float)Sq) : (1.f / s);
    }
  }

  const int wave = t >> 5, lane = t & 31;
  const int wm = wave & 1;         // 0..1 -> M offset
  const int wn = wave >> 1;        // 0..3 -> N offset (32 cols each)
  v8f c0 = {}, c1 = {};
  const _Float16* vbase = XP + (size_t)b * Sq * Dq + h * DA;

  for (int k0 = 0; k0 < Sq; k0 += 32) {
    __syncthreads();
    // V tile: 32 x 128 f16 staged via async global->LDS B128 (no VGPR data path)
    {
      const int kk = t >> 4;             // 0..15
      const int c8 = (t & 15) * 8;       // 0..120  (halfs; *2 = 16B aligned)
      #pragma unroll
      for (int i2 = 0; i2 < 2; ++i2) {
        const int k = kk + i2 * 16;
        const unsigned int ldsoff =
            (unsigned int)(uintptr_t)(vtile + k * DA + c8);
        const unsigned long long gaddr =
            (unsigned long long)(uintptr_t)(vbase + (size_t)(k0 + k) * Dq + c8);
        asm volatile("global_load_async_to_lds_b128 %0, %1, off"
                     :: "v"(ldsoff), "v"(gaddr) : "memory");
      }
    }
    // P tile: att weights already divided by denominator, f16
    {
      const int r  = t >> 3;             // 32 rows, 8 threads/row
      const int jj = (t & 7) * 4;
      const int i  = i0 + r;
      const int* arow = adj + ((size_t)(b * Sq + i)) * Sq + k0;
      if (k0 + 32 < Sq) __builtin_prefetch(arow + 32, 0, 0);  // next adjacency chunk
      const float m = rowm[r], dinv = rowdinv[r];
      const int uni = rowuni[r];
      #pragma unroll
      for (int q = 0; q < 4; ++q) {
        const int j = jj + q;
        float p;
        if (uni)       p = dinv;
        else if (arow[j] > 0 && s2m[k0 + j] > -1e29f)
                       p = __expf(s2m[k0 + j] - m) * dinv;
        else           p = 0.f;
        ptile[r * 32 + j] = (_Float16)p;
      }
    }
    // each wave drains its own async copies, then the barrier publishes vtile
    asm volatile("s_wait_asynccnt 0x0" ::: "memory");
    __syncthreads();

    const int Ml = lane & 15, half = lane >> 4;
    AF a;
    const _Float16* arow2 = ptile + (wm * 16 + Ml) * 32;
    #pragma unroll
    for (int e = 0; e < 16; ++e) {
      const int v = e >> 1;
      const int K = (v >> 2) * 16 + half * 8 + (v & 3) * 2 + (e & 1);
      a.h[e] = arow2[K];
    }
    AF b0, b1;
    #pragma unroll
    for (int e = 0; e < 16; ++e) {
      const int v = e >> 1;
      const int K = (v >> 2) * 16 + half * 8 + (v & 3) * 2 + (e & 1);
      b0.h[e] = vtile[K * DA + wn * 32 + Ml];
      b1.h[e] = vtile[K * DA + wn * 32 + 16 + Ml];
    }
    c0 = __builtin_amdgcn_wmma_f32_16x16x32_f16(false, a.v, false, b0.v, (short)0, c0, false, false);
    c1 = __builtin_amdgcn_wmma_f32_16x16x32_f16(false, a.v, false, b1.v, (short)0, c1, false, false);
  }

  // write out: OUT[b, i, h*128 + n]  (fp32)
  {
    CF cc0, cc1; cc0.v = c0; cc1.v = c1;
    const int nl = lane & 15, half = lane >> 4;
    #pragma unroll
    for (int r = 0; r < 8; ++r) {
      const int i = i0 + wm * 16 + half * 8 + r;
      float* orow = OUT + ((size_t)(b * Sq + i)) * Dq + h * DA + wn * 32;
      orow[nl]      = cc0.f[r];
      orow[16 + nl] = cc1.f[r];
    }
  }
}

// ---------------------------------------------------------------------------
extern "C" void kernel_launch(void* const* d_in, const int* in_sizes, int n_in,
                              void* d_out, int out_size, void* d_ws, size_t ws_size,
                              hipStream_t stream) {
  const float* x     = (const float*)d_in[0];   // [8,512,1024]
  const float* mask  = (const float*)d_in[1];   // [8,512]
  const int*   adjn  = (const int*)  d_in[2];   // [8,512,512]
  const float* W     = (const float*)d_in[3];   // [1024,1024]
  const float* att_w = (const float*)d_in[4];   // [256]
  float*       out   = (float*)d_out;           // [8,512,1024]

  _Float16* xp = (_Float16*)d_ws;                                 // 4096*1024 f16 = 8 MiB
  float*    s2 = (float*)((char*)d_ws + (size_t)4096 * 1024 * 2); // 8*8*512 f32

  // 1) projection GEMM (WMMA f16 -> f32 acc), store xp in f16
  proj_gemm<<<dim3(Dq / 64, (Bq * Sq) / 128), 256, 0, stream>>>(x, W, xp);

  // 2) per-(b,h,j) score component s2 (s1/bias cancel in softmax)
  s2_kernel<<<(Bq * Hq * Sq) / 256, 256, 0, stream>>>(xp, att_w, s2);

  // 3) fused masked softmax + att@V (WMMA + async LDS staging)
  attn_kernel<<<dim3(Sq / 32, Hq, Bq), 256, 0, stream>>>(adjn, mask, s2, xp, out);
}
